// EncoderProcesserDecoder_45509473468805
// MI455X (gfx1250) — matrix operand order repaired and checked by reading, not compiled, not run on hardware
//
#include <hip/hip_runtime.h>

#define HDIM   64
#define MPR    15
#define NNODES 100000
#define E0E    600000
#define ETE    1200000

typedef __attribute__((ext_vector_type(16))) __bf16 v16bf;
typedef __attribute__((ext_vector_type(8)))  __bf16 v8bf;
typedef __attribute__((ext_vector_type(8)))  float  v8f;

struct MlpDev {
  const __bf16 *w1t, *w2t, *w3t, *w4t;       // bf16 weights, N-major [N][K]
  const float  *b1, *b2, *b3, *b4, *g, *be;  // fp32 biases + LN params
};

// ---------------------------------------------------------------------------
// WMMA fragment loaders (wave32, 16x16x32 bf16)
// A 16x32 layout: lanes 0-15 -> M=0..15 / K sub-blocks {0..7,16..23};
//                 lanes 16-31 -> same M / K {8..15,24..31}.
// B 32x16 layout: lanes 0-15 -> N / K {0..7,16..23}; lanes 16-31 -> N / K {8..15,24..31}.
// ---------------------------------------------------------------------------
__device__ inline v16bf load_a_frag(const __bf16* rowp, int k0, int half) {
  const __bf16* q = rowp + k0 + half * 8;
  v8bf lo = *(const v8bf*)(q);
  v8bf hi = *(const v8bf*)(q + 16);
  v16bf f;
#pragma unroll
  for (int i = 0; i < 8; ++i) { f[i] = lo[i]; f[i + 8] = hi[i]; }
  return f;
}

__device__ inline v16bf load_b_frag(const __bf16* rowp, int k0, int half) {
  const __bf16* q = rowp + k0 + half * 8;
  v8bf lo = *(const v8bf*)(q);
  v8bf hi = *(const v8bf*)(q + 16);
  v16bf f;
#pragma unroll
  for (int i = 0; i < 8; ++i) { f[i] = lo[i]; f[i + 8] = hi[i]; }
  return f;
}

// One dense layer for a 64x64 tile: 4 waves, wave w owns rows [16w,16w+16).
// sIn: LDS activations [64][KC*32] bf16. wt: global bf16 [NT*16][KC*32] N-major.
template<int KC, int NT, bool RELU, bool TOF32>
__device__ inline void gemm_layer(const __bf16* sIn, int ldin,
                                  const __bf16* __restrict__ wt,
                                  const float* __restrict__ bias, int biasN,
                                  __bf16* outB, float* outF,
                                  int w, int lane)
{
  const int half = lane >> 4;
  const int l16  = lane & 15;
  const __bf16* arow = sIn + (size_t)(w * 16 + l16) * ldin;
  v16bf af[KC];
#pragma unroll
  for (int kc = 0; kc < KC; ++kc) af[kc] = load_a_frag(arow, 32 * kc, half);
#pragma unroll
  for (int nt = 0; nt < NT; ++nt) {
    const int n = nt * 16 + l16;
    const __bf16* brow = wt + (size_t)n * (KC * 32);
    v8f acc = {0.f, 0.f, 0.f, 0.f, 0.f, 0.f, 0.f, 0.f};
#pragma unroll
    for (int kc = 0; kc < KC; ++kc) {
      v16bf bfv = load_b_frag(brow, 32 * kc, half);
      acc = __builtin_amdgcn_wmma_f32_16x16x32_bf16(false, af[kc], false, bfv,
                                                    (short)0, acc, false, false);
    }
    const float bn = (n < biasN) ? bias[n] : 0.f;
#pragma unroll
    for (int r = 0; r < 8; ++r) {
      const int m = w * 16 + half * 8 + r;   // D layout: M = r + 8*half
      float v = acc[r] + bn;
      if (RELU) v = fmaxf(v, 0.f);
      if (TOF32) outF[m * 64 + n] = v;
      else       outB[m * 64 + n] = (__bf16)v;
    }
  }
}

// 4-layer MLP + LayerNorm for a 64-row tile. sA >= 64*max(KC1*32,64) bf16.
template<int KC1, int NT4>
__device__ void mlp4_ln(__bf16* sA, __bf16* sTmp, float* sOut,
                        const MlpDev& mp, int nvalid)
{
  const int tid  = threadIdx.x;
  const int w    = tid >> 5;
  const int lane = tid & 31;

  gemm_layer<KC1, 4, true,  false>(sA,   KC1 * 32, mp.w1t, mp.b1, 64,     sTmp, nullptr, w, lane);
  __syncthreads();
  gemm_layer<2,   4, true,  false>(sTmp, 64,       mp.w2t, mp.b2, 64,     sA,   nullptr, w, lane);
  __syncthreads();
  gemm_layer<2,   4, true,  false>(sA,   64,       mp.w3t, mp.b3, 64,     sTmp, nullptr, w, lane);
  __syncthreads();
  gemm_layer<2, NT4, false, true >(sTmp, 64,       mp.w4t, mp.b4, nvalid, nullptr, sOut, w, lane);
  __syncthreads();

  if (tid < 64) {
    float* row = sOut + tid * 64;
    float s = 0.f, ss = 0.f;
    for (int k = 0; k < nvalid; ++k) { float v = row[k]; s += v; ss += v * v; }
    const float invn = 1.f / (float)nvalid;
    const float mu  = s * invn;
    const float var = ss * invn - mu * mu;
    const float rs  = rsqrtf(var + 1e-5f);
    for (int k = 0; k < nvalid; ++k)
      row[k] = (row[k] - mu) * rs * mp.g[k] + mp.be[k];
  }
  __syncthreads();
}

// ---------------------------------------------------------------------------
// Utility kernels
// ---------------------------------------------------------------------------
__global__ __launch_bounds__(256) void prep_wt_kernel(const float* __restrict__ W,
    __bf16* __restrict__ dst, int din, int dout, int kpad, int npad)
{
  int i = blockIdx.x * 256 + threadIdx.x;
  if (i >= kpad * npad) return;
  int n = i / kpad;
  int k = i - n * kpad;
  float v = (k < din && n < dout) ? W[(size_t)k * dout + n] : 0.f;
  dst[i] = (__bf16)v;   // dst layout: [npad][kpad], N-major (WMMA B source)
}

__global__ __launch_bounds__(256) void zero_f32_kernel(float* __restrict__ p,
                                                       unsigned long long n)
{
  unsigned long long i = (unsigned long long)blockIdx.x * 256ull + threadIdx.x;
  if (i < n) p[i] = 0.f;
}

// ---------------------------------------------------------------------------
// Edge kernels. MODE 0: encoder (x_edge 9->pad32, scatter agg, E0 edges)
//              MODE 1: enc_eb1  ([xn[s]|xn[d]]=128 -> xe, ET edges)
//              MODE 2: GN edge  ([xn[s]|xn[d]|xe]=192 -> scatter agg, xe+=, ET)
//              MODE 3: decoder  ([xn[s]|xn[d]]=128 -> out[.,3], E0 edges)
// ---------------------------------------------------------------------------
template<int MODE>
__global__ __launch_bounds__(128) void edge_kernel(
    MlpDev mp,
    const float* __restrict__ x_edge,
    const int*   __restrict__ eidx,     // [2][ET] row-major: s then d
    const float* __restrict__ xn,
    float*       __restrict__ xe,
    float*       __restrict__ agg,
    float*       __restrict__ out,
    int nE, int eTotal)
{
  constexpr int KIN = (MODE == 0) ? 32 : (MODE == 2) ? 192 : 128;
  constexpr int SAE = 64 * (KIN < 64 ? 64 : KIN);
  __shared__ __bf16 sA[SAE];
  __shared__ __bf16 sTmp[64 * 64];
  __shared__ float  sOut[64 * 64];
  __shared__ int    sS[64], sD[64];

  const int tid  = threadIdx.x;
  const int base = blockIdx.x * 64;

  if (tid < 64) {
    int e = base + tid;
    sS[tid] = (e < nE) ? eidx[e] : 0;
    sD[tid] = (e < nE) ? eidx[eTotal + e] : 0;
  }
  __syncthreads();

  // Gather -> bf16 LDS tile
  for (int i = tid; i < 64 * KIN; i += 128) {
    int m = i / KIN;
    int k = i - m * KIN;
    int e = base + m;
    float v = 0.f;
    if (e < nE) {
      if constexpr (MODE == 0) {
        if (k < 9) v = x_edge[(size_t)e * 9 + k];
      } else {
        if (k < 64)       v = xn[(size_t)sS[m] * 64 + k];
        else if (k < 128) v = xn[(size_t)sD[m] * 64 + (k - 64)];
        else              v = xe[(size_t)e * 64 + (k - 128)];
      }
    }
    sA[m * KIN + k] = (__bf16)v;
  }
  __syncthreads();

  mlp4_ln<KIN / 32, (MODE == 3 ? 1 : 4)>(sA, sTmp, sOut, mp, (MODE == 3 ? 3 : 64));

  if constexpr (MODE == 0 || MODE == 2) {
    for (int i = tid; i < 64 * 64; i += 128) {
      int m = i >> 6, k = i & 63;
      int e = base + m;
      if (e < nE) {
        float v = sOut[i];
        atomicAdd(&agg[(size_t)sS[m] * 64 + k], v);
        atomicAdd(&agg[(size_t)sD[m] * 64 + k], v);
        if constexpr (MODE == 2) xe[(size_t)e * 64 + k] += v;  // residual
      }
    }
  } else if constexpr (MODE == 1) {
    for (int i = tid; i < 64 * 64; i += 128) {
      int m = i >> 6, k = i & 63;
      int e = base + m;
      if (e < nE) xe[(size_t)e * 64 + k] = sOut[i];
    }
  } else {  // MODE 3: decoder output, 3 cols
    for (int i = tid; i < 64 * 3; i += 128) {
      int m = i / 3, k = i - m * 3;
      int e = base + m;
      if (e < nE) out[(size_t)e * 3 + k] = sOut[m * 64 + k];
    }
  }
}

// Node kernels. MODE 0: enc_nb (agg=64 -> xn). MODE 1: GN node ([xn|agg]=128, xn+=).
// Both zero agg after consuming it.
template<int MODE>
__global__ __launch_bounds__(128) void node_kernel(
    MlpDev mp, float* __restrict__ xn, float* __restrict__ agg, int nN)
{
  constexpr int KIN = (MODE == 0) ? 64 : 128;
  __shared__ __bf16 sA[64 * KIN];
  __shared__ __bf16 sTmp[64 * 64];
  __shared__ float  sOut[64 * 64];
  const int tid  = threadIdx.x;
  const int base = blockIdx.x * 64;

  for (int i = tid; i < 64 * KIN; i += 128) {
    int m = i / KIN, k = i - m * KIN;
    int n = base + m;
    float v = 0.f;
    if (n < nN) {
      if constexpr (MODE == 0) v = agg[(size_t)n * 64 + k];
      else v = (k < 64) ? xn[(size_t)n * 64 + k] : agg[(size_t)n * 64 + (k - 64)];
    }
    sA[m * KIN + k] = (__bf16)v;
  }
  __syncthreads();

  for (int i = tid; i < 64 * 64; i += 128) {     // zero agg for next round
    int n = base + (i >> 6);
    if (n < nN) agg[(size_t)n * 64 + (i & 63)] = 0.f;
  }

  mlp4_ln<KIN / 32, 4>(sA, sTmp, sOut, mp, 64);

  for (int i = tid; i < 64 * 64; i += 128) {
    int n = base + (i >> 6), k = i & 63;
    if (n < nN) {
      if constexpr (MODE == 0) xn[(size_t)n * 64 + k]  = sOut[i];
      else                     xn[(size_t)n * 64 + k] += sOut[i];
    }
  }
}

// ---------------------------------------------------------------------------
// Host side
// ---------------------------------------------------------------------------
struct MlpJob {
  MlpDev dev;
  const float* W[4];
  int din1, kpad1, dout4, npad4;
};

static inline size_t align256(size_t x) { return (x + 255) & ~(size_t)255; }

// params tree flatten: each MLP dict sorted -> W1 W2 W3 W4 b1 b2 b3 b4 be g
static void carve_mlp(MlpJob& j, void* const* d_in, int base,
                      int din1, int kpad1, int dout4, int npad4, char*& cur)
{
  for (int i = 0; i < 4; ++i) j.W[i] = (const float*)d_in[base + i];
  j.dev.b1 = (const float*)d_in[base + 4];
  j.dev.b2 = (const float*)d_in[base + 5];
  j.dev.b3 = (const float*)d_in[base + 6];
  j.dev.b4 = (const float*)d_in[base + 7];
  j.dev.be = (const float*)d_in[base + 8];
  j.dev.g  = (const float*)d_in[base + 9];
  j.din1 = din1; j.kpad1 = kpad1; j.dout4 = dout4; j.npad4 = npad4;
  auto carve = [&](size_t elems) {
    __bf16* p = (__bf16*)cur; cur += align256(elems * sizeof(__bf16)); return p;
  };
  j.dev.w1t = carve((size_t)64 * kpad1);
  j.dev.w2t = carve(64 * 64);
  j.dev.w3t = carve(64 * 64);
  j.dev.w4t = carve((size_t)npad4 * 64);
}

static void launch_preps(const MlpJob& j, hipStream_t stream)
{
  auto go = [&](const float* W, const __bf16* dst, int din, int dout, int kpad, int npad) {
    int n = kpad * npad;
    prep_wt_kernel<<<dim3((n + 255) / 256), dim3(256), 0, stream>>>(
        W, (__bf16*)dst, din, dout, kpad, npad);
  };
  go(j.W[0], j.dev.w1t, j.din1, 64, j.kpad1, 64);
  go(j.W[1], j.dev.w2t, 64, 64, 64, 64);
  go(j.W[2], j.dev.w3t, 64, 64, 64, 64);
  go(j.W[3], j.dev.w4t, 64, j.dout4, 64, j.npad4);
}

extern "C" void kernel_launch(void* const* d_in, const int* in_sizes, int n_in,
                              void* d_out, int out_size, void* d_ws, size_t ws_size,
                              hipStream_t stream)
{
  (void)in_sizes; (void)n_in; (void)out_size;
  const float* x_edge = (const float*)d_in[0];
  const int*   eidx   = (const int*)d_in[1];

  char* cur = (char*)d_ws;

  // params subtree (sorted keys): dec, enc_eb, enc_eb1, enc_nb, gn[0..14]{eb,nb}
  MlpJob dec, enc_eb, enc_eb1, enc_nb;
  MlpJob gne[MPR], gnn[MPR];
  carve_mlp(dec,     d_in,  2, 128, 128,  3, 16, cur);
  carve_mlp(enc_eb,  d_in, 12,   9,  32, 64, 64, cur);
  carve_mlp(enc_eb1, d_in, 22, 128, 128, 64, 64, cur);
  carve_mlp(enc_nb,  d_in, 32,  64,  64, 64, 64, cur);
  for (int i = 0; i < MPR; ++i) {
    carve_mlp(gne[i], d_in, 42 + 20 * i,      192, 192, 64, 64, cur);
    carve_mlp(gnn[i], d_in, 42 + 20 * i + 10, 128, 128, 64, 64, cur);
  }

  const int    tilesN = (NNODES + 63) / 64;
  const size_t padN   = (size_t)tilesN * 64;
  float* xn  = (float*)cur; cur += align256(padN * 64 * sizeof(float));
  float* agg = (float*)cur; cur += align256(padN * 64 * sizeof(float));
  float* xe  = (float*)cur; cur += align256((size_t)ETE * 64 * sizeof(float));

  if ((size_t)(cur - (char*)d_ws) > ws_size) return;  // deterministic guard

  launch_preps(dec, stream);
  launch_preps(enc_eb, stream);
  launch_preps(enc_eb1, stream);
  launch_preps(enc_nb, stream);
  for (int i = 0; i < MPR; ++i) { launch_preps(gne[i], stream); launch_preps(gnn[i], stream); }

  {
    unsigned long long n = (unsigned long long)padN * 64;
    zero_f32_kernel<<<dim3((unsigned)((n + 255) / 256)), dim3(256), 0, stream>>>(agg, n);
  }

  const int tilesE0 = (E0E + 63) / 64;
  const int tilesET = (ETE + 63) / 64;

  // Encoder: edge MLP + segment sums into agg, then node MLP -> xn
  edge_kernel<0><<<dim3(tilesE0), dim3(128), 0, stream>>>(
      enc_eb.dev, x_edge, eidx, xn, xe, agg, (float*)d_out, E0E, ETE);
  node_kernel<0><<<dim3(tilesN), dim3(128), 0, stream>>>(enc_nb.dev, xn, agg, NNODES);
  edge_kernel<1><<<dim3(tilesET), dim3(128), 0, stream>>>(
      enc_eb1.dev, x_edge, eidx, xn, xe, agg, (float*)d_out, ETE, ETE);

  // 15 message-passing rounds
  for (int i = 0; i < MPR; ++i) {
    edge_kernel<2><<<dim3(tilesET), dim3(128), 0, stream>>>(
        gne[i].dev, x_edge, eidx, xn, xe, agg, (float*)d_out, ETE, ETE);
    node_kernel<1><<<dim3(tilesN), dim3(128), 0, stream>>>(gnn[i].dev, xn, agg, NNODES);
  }

  // Decoder -> d_out [E0][3] fp32
  edge_kernel<3><<<dim3(tilesE0), dim3(128), 0, stream>>>(
      dec.dev, x_edge, eidx, xn, xe, agg, (float*)d_out, E0E, ETE);
}
